// SoftEditDistanceLoss_4964982194563
// MI455X (gfx1250) — matrix operand order repaired and checked by reading, not compiled, not run on hardware
//
#include <hip/hip_runtime.h>
#include <hip/hip_bf16.h>

#define LEN1 8192
#define LEN2 8192
#define ALPH 32

typedef __attribute__((ext_vector_type(16))) __bf16 v16bf;
typedef __attribute__((ext_vector_type(8)))  __bf16 v8bf;
typedef __attribute__((ext_vector_type(8)))  float  v8f;

// ---------------------------------------------------------------------------
// fp32 -> bf16 conversion of both sequences (2 MB total; L2-resident after).
// ---------------------------------------------------------------------------
__global__ __launch_bounds__(256) void cvt_pair_kernel(
    const float* __restrict__ s1, const float* __restrict__ s2,
    __bf16* __restrict__ o1, __bf16* __restrict__ o2, int n) {
  int i = blockIdx.x * blockDim.x + threadIdx.x;
  if (i < n) {
    o1[i] = (__bf16)s1[i];
    o2[i] = (__bf16)s2[i];
  }
}

__global__ void zero_accum_kernel(float* accum) {
  if (threadIdx.x < 2) accum[threadIdx.x] = 0.0f;
}

// Build the 16-element bf16 per-lane WMMA operand from two contiguous 8-elem
// (16-byte) chunks -> two global_load_b128 per lane.
__device__ __forceinline__ v16bf load_tile16(const __bf16* p0, const __bf16* p1) {
  v8bf lo = *(const v8bf*)p0;
  v8bf hi = *(const v8bf*)p1;
  return __builtin_shufflevector(lo, hi, 0, 1, 2, 3, 4, 5, 6, 7,
                                 8, 9, 10, 11, 12, 13, 14, 15);
}

__device__ __forceinline__ v8f wmma_bf16(v16bf a, v16bf b) {
  v8f c = {};
  return __builtin_amdgcn_wmma_f32_16x16x32_bf16(
      /*neg_a=*/false, a, /*neg_b=*/false, b,
      /*c_mod=*/(short)0, c, /*reuse_a=*/false, /*reuse_b=*/false);
}

// ---------------------------------------------------------------------------
// One block = 16 complete rows of the similarity matrix (all nCols columns).
// 8 waves split the column tiles, 2 tiles (2 independent WMMAs) per loop
// iteration so the WMMA->VALU hazard window of one tile is covered by the
// other tile's issue stream. Pass 1: exact row max (sim >= 0). Pass 2:
// recompute tiles, accumulate sum(exp) and sum(exp*sim) per row.
// Block emits one atomicAdd of sum_i( ws_i / sum_i ) into *accum.
// Called twice with (A,B) swapped to get row- and col-softmax scores.
// ---------------------------------------------------------------------------
__global__ __launch_bounds__(256) void softmax_strip_kernel(
    const __bf16* __restrict__ A, const __bf16* __restrict__ B,
    const float* __restrict__ tempPtr, float* __restrict__ accum, int nCols) {
  __shared__ int   sMaxBits[16];
  __shared__ float sSum[16];
  __shared__ float sWS[16];

  const int lane    = threadIdx.x & 31;
  const int wave    = threadIdx.x >> 5;   // 0..7
  const int half    = lane >> 4;          // 0 or 1
  const int mr      = lane & 15;
  const int rowBase = blockIdx.x * 16;
  const float invT  = 1.0f / tempPtr[0];

  if (threadIdx.x < 16) {
    sMaxBits[threadIdx.x] = 0;      // sim >= 0, so 0 is a safe -inf surrogate
    sSum[threadIdx.x] = 0.0f;
    sWS[threadIdx.x]  = 0.0f;
  }
  __syncthreads();

  // A tile (16 x 32 bf16), contiguous K order: lane<16 -> M=lane, K={0..15};
  // lane>=16 -> M=lane-16, K={16..31}.
  const __bf16* arow = A + (size_t)(rowBase + mr) * ALPH + half * 16;
  const v16bf a = load_tile16(arow, arow + 8);

  // Per-lane B row pointer for column tile base 0 (advance by tiles).
  const __bf16* bbase = B + (size_t)mr * ALPH + half * 16;

  // ---------------- Pass 1: per-row maxima ----------------
  float rmax[8];
#pragma unroll
  for (int r = 0; r < 8; ++r) rmax[r] = 0.0f;

  for (int n0 = wave * 32; n0 < nCols; n0 += 32 * 8) {
    const __bf16* brow0 = bbase + (size_t)n0 * ALPH;
    const __bf16* brow1 = brow0 + 16 * ALPH;
    v16bf b0 = load_tile16(brow0, brow0 + 8);
    v16bf b1 = load_tile16(brow1, brow1 + 8);
    v8f c0 = wmma_bf16(a, b0);
    v8f c1 = wmma_bf16(a, b1);
#pragma unroll
    for (int r = 0; r < 8; ++r)
      rmax[r] = fmaxf(fmaxf(rmax[r], c0[r]), c1[r]);
  }
  // Reduce over the 16 lanes holding the same row set (xor masks stay in-half).
#pragma unroll
  for (int r = 0; r < 8; ++r) {
#pragma unroll
    for (int m = 8; m >= 1; m >>= 1)
      rmax[r] = fmaxf(rmax[r], __shfl_xor(rmax[r], m, 32));
  }
  if (mr == 0) {  // lanes 0 and 16
#pragma unroll
    for (int r = 0; r < 8; ++r)
      atomicMax(&sMaxBits[r + half * 8], __float_as_int(rmax[r]));
  }
  __syncthreads();

  // ---------------- Pass 2: exp sums against true row max ----------------
  float rowm[8], sume[8], wsum[8];
#pragma unroll
  for (int r = 0; r < 8; ++r) {
    rowm[r] = __int_as_float(sMaxBits[r + half * 8]);
    sume[r] = 0.0f;
    wsum[r] = 0.0f;
  }

  for (int n0 = wave * 32; n0 < nCols; n0 += 32 * 8) {
    const __bf16* brow0 = bbase + (size_t)n0 * ALPH;
    const __bf16* brow1 = brow0 + 16 * ALPH;
    v16bf b0 = load_tile16(brow0, brow0 + 8);
    v16bf b1 = load_tile16(brow1, brow1 + 8);
    v8f c0 = wmma_bf16(a, b0);
    v8f c1 = wmma_bf16(a, b1);
#pragma unroll
    for (int r = 0; r < 8; ++r) {
      float s0 = c0[r];
      float s1 = c1[r];
      float e0 = __expf((s0 - rowm[r]) * invT);
      float e1 = __expf((s1 - rowm[r]) * invT);
      sume[r] += e0 + e1;
      wsum[r] += e0 * s0 + e1 * s1;
    }
  }
#pragma unroll
  for (int r = 0; r < 8; ++r) {
#pragma unroll
    for (int m = 8; m >= 1; m >>= 1) {
      sume[r] += __shfl_xor(sume[r], m, 32);
      wsum[r] += __shfl_xor(wsum[r], m, 32);
    }
  }
  if (mr == 0) {
#pragma unroll
    for (int r = 0; r < 8; ++r) {
      atomicAdd(&sSum[r + half * 8], sume[r]);   // ds_add_f32
      atomicAdd(&sWS[r + half * 8],  wsum[r]);
    }
  }
  __syncthreads();

  // Block partial: sum over its 16 rows of (weighted sum / partition).
  if (wave == 0) {
    float p = (lane < 16) ? (sWS[lane] / sSum[lane]) : 0.0f;
#pragma unroll
    for (int m = 16; m >= 1; m >>= 1) p += __shfl_xor(p, m, 32);
    if (lane == 0) atomicAdd(accum, p);          // global_atomic_add_f32
  }
}

__global__ void finalize_kernel(const float* __restrict__ accum,
                                float* __restrict__ out) {
  if (threadIdx.x == 0) {
    float msr = accum[0];
    float msc = accum[1];
    float dist_row = (float)LEN1 - msr;
    float dist_col = (float)LEN2 - msc;
    float total = fmaxf((dist_row + dist_col) * 0.5f, 0.0f);
    float len_pen = fabsf((float)(LEN1 - LEN2));
    out[0] = total + len_pen;
  }
}

// ---------------------------------------------------------------------------
extern "C" void kernel_launch(void* const* d_in, const int* in_sizes, int n_in,
                              void* d_out, int out_size, void* d_ws, size_t ws_size,
                              hipStream_t stream) {
  (void)in_sizes; (void)n_in; (void)out_size; (void)ws_size;

  const float* seq1 = (const float*)d_in[0];
  const float* seq2 = (const float*)d_in[1];
  const float* temp = (const float*)d_in[2];
  float* out = (float*)d_out;

  // Workspace layout: [seq1 bf16 | seq2 bf16 | accum[2] f32]
  __bf16* s1b = (__bf16*)d_ws;
  __bf16* s2b = (__bf16*)((char*)d_ws + (size_t)LEN1 * ALPH * sizeof(__bf16));
  float* accum = (float*)((char*)d_ws +
                          (size_t)(LEN1 + LEN2) * ALPH * sizeof(__bf16));

  const int n = LEN1 * ALPH;  // == LEN2 * ALPH
  cvt_pair_kernel<<<(n + 255) / 256, 256, 0, stream>>>(seq1, seq2, s1b, s2b, n);
  zero_accum_kernel<<<1, 32, 0, stream>>>(accum);

  // Row-softmax direction: rows of sim.
  softmax_strip_kernel<<<LEN1 / 16, 256, 0, stream>>>(s1b, s2b, temp,
                                                      accum + 0, LEN2);
  // Col-softmax direction == row pass on sim^T (operands swapped).
  softmax_strip_kernel<<<LEN2 / 16, 256, 0, stream>>>(s2b, s1b, temp,
                                                      accum + 1, LEN1);

  finalize_kernel<<<1, 32, 0, stream>>>(accum, out);
}
